// GraphGATClassifier_27608049779521
// MI455X (gfx1250) — matrix operand order repaired and checked by reading, not compiled, not run on hardware
//
#include <hip/hip_runtime.h>

#define NN 100000
#define EE 1600000
#define GG 256
#define INF_ 20
#define HH 128
#define CC 26
#define EPS 1e-5f
#define NPAD 100032   // ceil(N/64)*64

typedef __attribute__((ext_vector_type(16))) __bf16 bf16x16;
typedef __attribute__((ext_vector_type(8)))  float  floatx8;

// ---------------------------------------------------------------- degree / count
__global__ __launch_bounds__(256) void k_deg(const int* __restrict__ dst, float* deg) {
    int e = blockIdx.x * 256 + threadIdx.x;
    if (e < EE) atomicAdd(&deg[dst[e]], 1.0f);
}
__global__ __launch_bounds__(256) void k_cnt(const int* __restrict__ ng, float* cnt) {
    int n = blockIdx.x * 256 + threadIdx.x;
    if (n < NN) atomicAdd(&cnt[ng[n]], 1.0f);
}

// ---------------------------------------------------------------- input LayerNorm
__global__ __launch_bounds__(256) void k_ln(const float* __restrict__ feat,
                                            const float* __restrict__ g,
                                            const float* __restrict__ b,
                                            float* __restrict__ hln) {
    int n = blockIdx.x * 256 + threadIdx.x;
    if (n >= NN) return;
    float v[INF_];
    float mu = 0.f;
    #pragma unroll
    for (int i = 0; i < INF_; ++i) { v[i] = feat[n * INF_ + i]; mu += v[i]; }
    mu *= (1.0f / INF_);
    float var = 0.f;
    #pragma unroll
    for (int i = 0; i < INF_; ++i) { float d = v[i] - mu; var += d * d; }
    var *= (1.0f / INF_);
    float is = rsqrtf(var + EPS);
    #pragma unroll
    for (int i = 0; i < INF_; ++i)
        hln[n * 32 + i] = (v[i] - mu) * is * g[i] + b[i];
}

// ---------------------------------------------------------------- edge scatter-add
template <int CH, int ST>
__global__ __launch_bounds__(256) void k_edge_agg(const int* __restrict__ src,
                                                  const int* __restrict__ dst,
                                                  const float* __restrict__ h,
                                                  float* __restrict__ agg) {
    int gid = blockIdx.x * 256 + threadIdx.x;
    if (gid >= EE * CH) return;
    int e = gid / CH;
    int c = gid - e * CH;
    const float4 v = *(const float4*)(h + (size_t)src[e] * ST + c * 4);
    float* a = agg + (size_t)dst[e] * ST + c * 4;
    atomicAdd(a + 0, v.x); atomicAdd(a + 1, v.y);
    atomicAdd(a + 2, v.z); atomicAdd(a + 3, v.w);
}

// ---------------------------------------------------------------- bf16 A-matrix packing
// layer0: A0[n][64] = [ hln(20) | 0(12) | agg*inv_deg(20) | 0(12) ]
__global__ __launch_bounds__(256) void k_pack0(const float* __restrict__ hln,
                                               const float* __restrict__ agg,
                                               const float* __restrict__ deg,
                                               __bf16* __restrict__ A0) {
    int gid = blockIdx.x * 256 + threadIdx.x;
    int n = gid >> 6, k = gid & 63;
    if (n >= NPAD) return;
    float v = 0.f;
    if (n < NN) {
        if (k < INF_) v = hln[n * 32 + k];
        else if (k >= 32 && k < 32 + INF_) {
            float id = 1.0f / fmaxf(deg[n], 1.0f);
            v = agg[n * 32 + (k - 32)] * id;
        }
    }
    A0[(size_t)n * 64 + k] = (__bf16)v;
}
// layer1: A1[n][256] = [ h(128) | agg*inv_deg(128) ]
__global__ __launch_bounds__(256) void k_pack1(const float* __restrict__ h,
                                               const float* __restrict__ agg,
                                               const float* __restrict__ deg,
                                               __bf16* __restrict__ A1) {
    int gid = blockIdx.x * 256 + threadIdx.x;
    int n = gid >> 8, k = gid & 255;
    if (n >= NPAD) return;
    float v = 0.f;
    if (n < NN) {
        if (k < HH) v = h[(size_t)n * HH + k];
        else {
            float id = 1.0f / fmaxf(deg[n], 1.0f);
            v = agg[(size_t)n * HH + (k - HH)] * id;
        }
    }
    A1[(size_t)n * 256 + k] = (__bf16)v;
}

// ---------------------------------------------------------------- weight prep (transposed, concat-K, bf16)
__global__ __launch_bounds__(256) void k_prep_w0(const float* __restrict__ ws,
                                                 const float* __restrict__ wn,
                                                 __bf16* __restrict__ WT) {
    int gid = blockIdx.x * 256 + threadIdx.x;   // 128*64
    int n = gid >> 6, k = gid & 63;
    float v = 0.f;
    if (k < INF_)                      v = ws[k * HH + n];
    else if (k >= 32 && k < 32 + INF_) v = wn[(k - 32) * HH + n];
    WT[n * 64 + k] = (__bf16)v;
}
__global__ __launch_bounds__(256) void k_prep_w1(const float* __restrict__ ws,
                                                 const float* __restrict__ wn,
                                                 __bf16* __restrict__ WT) {
    int gid = blockIdx.x * 256 + threadIdx.x;   // 128*256
    int n = gid >> 8, k = gid & 255;
    float v = (k < HH) ? ws[k * HH + n] : wn[(k - HH) * HH + n];
    WT[n * 256 + k] = (__bf16)v;
}

// ---------------------------------------------------------------- WMMA SAGE GEMM
// out[m][0..128) = relu( A[m][:] @ WT^T + bias ); 64 rows/block, 8 waves.
// Output buffer is padded to NPAD rows -> branch-free epilogue.
template <int KTILES>
__global__ __launch_bounds__(256) void sage_gemm_kernel(const __bf16* __restrict__ A,
                                                        const __bf16* __restrict__ WT,
                                                        const float* __restrict__ bias,
                                                        float* __restrict__ out) {
    constexpr int K = KTILES * 32;
    constexpr int LDSS = K + 8;                 // row stride in bf16 (16B aligned, conflict-free)
    extern __shared__ __align__(16) __bf16 sA[];  // [64][LDSS]
    const int tid  = threadIdx.x;
    const int lane = tid & 31;
    const int wave = tid >> 5;
    const int m0   = blockIdx.x * 64;

    // ---- async stage of A tile into LDS (GLOBAL_LOAD_ASYNC_TO_LDS_B128, ASYNCcnt)
    constexpr int VPR = K / 8;                  // 16B chunks per row
    #pragma unroll
    for (int i = 0; i < (64 * VPR) / 256; ++i) {
        int idx = tid + i * 256;
        int r = idx / VPR, cv = idx - r * VPR;
        const __bf16* gaddr = A + (size_t)(m0 + r) * K + cv * 8;
        unsigned ldsoff = (unsigned)(unsigned long long)(sA + r * LDSS + cv * 8);
        asm volatile("global_load_async_to_lds_b128 %0, %1, off"
                     :: "v"(ldsoff), "v"(gaddr) : "memory");
    }
    asm volatile("s_wait_asynccnt 0x0" ::: "memory");
    __syncthreads();

    const int n0 = wave * 16;
    const int nl = lane & 15;
    floatx8 acc[4];
    const floatx8 zero = {0.f, 0.f, 0.f, 0.f, 0.f, 0.f, 0.f, 0.f};
    #pragma unroll
    for (int mt = 0; mt < 4; ++mt) acc[mt] = zero;

    const __bf16* wrow = WT + (size_t)(n0 + nl) * K;
    const int hi = (lane < 16) ? 0 : 1;

    #pragma unroll
    for (int ks = 0; ks < KTILES; ++ks) {
        // B fragment: lanes 0-15 K=0..15, lanes 16-31 K=16..31 (contiguous)
        union { bf16x16 v; uint4 q[2]; } bfr;
        const int kb_b = ks * 32 + hi * 16;
        bfr.q[0] = *(const uint4*)(wrow + kb_b);
        bfr.q[1] = *(const uint4*)(wrow + kb_b + 8);
        // A fragments for all 4 m-tiles first, so ds latency overlaps the WMMA chain
        const int kb_a = ks * 32 + hi * 16;     // lo half K=0..15, hi half K=16..31 (contiguous)
        union { bf16x16 v; uint4 q[2]; } afr[4];
        #pragma unroll
        for (int mt = 0; mt < 4; ++mt) {
            const __bf16* arow = sA + (mt * 16 + nl) * LDSS;
            afr[mt].q[0] = *(const uint4*)(arow + kb_a);
            afr[mt].q[1] = *(const uint4*)(arow + kb_a + 8);
        }
        #pragma unroll
        for (int mt = 0; mt < 4; ++mt)
            acc[mt] = __builtin_amdgcn_wmma_f32_16x16x32_bf16(
                false, afr[mt].v, false, bfr.v, (short)0, acc[mt], false, false);
    }

    // epilogue: bias + relu; C layout: VGPR r -> M = r + 8*hi, N = lane&15
    const int col = n0 + nl;
    const float bv = bias[col];
    const int rbase = m0 + hi * 8;
    #pragma unroll
    for (int mt = 0; mt < 4; ++mt) {
        #pragma unroll
        for (int r = 0; r < 8; ++r) {
            float v = acc[mt][r] + bv;
            out[(size_t)(rbase + mt * 16 + r) * HH + col] = v > 0.f ? v : 0.f;
        }
    }
}

// ---------------------------------------------------------------- GraphNorm
__global__ __launch_bounds__(256) void k_gn_sum(const float* __restrict__ h,
                                                const int* __restrict__ ng,
                                                float* __restrict__ gsum,
                                                float* __restrict__ gsum2) {
    int gid = blockIdx.x * 256 + threadIdx.x;
    if (gid >= NN * HH) return;
    int n = gid >> 7, f = gid & 127;
    float v = h[gid];
    int g = ng[n];
    atomicAdd(&gsum[g * HH + f], v);
    atomicAdd(&gsum2[g * HH + f], v * v);
}
__global__ __launch_bounds__(256) void k_gn_apply(float* __restrict__ h,
                                                  const int* __restrict__ ng,
                                                  const float* __restrict__ cnt,
                                                  const float* __restrict__ gsum,
                                                  const float* __restrict__ gsum2,
                                                  const float* __restrict__ gg,
                                                  const float* __restrict__ gb,
                                                  const float* __restrict__ ga) {
    int gid = blockIdx.x * 256 + threadIdx.x;
    if (gid >= NN * HH) return;
    int n = gid >> 7, f = gid & 127;
    int g = ng[n];
    float ic = 1.0f / fmaxf(cnt[g], 1.0f);
    float m  = gsum[g * HH + f] * ic;
    float m2 = gsum2[g * HH + f] * ic;
    float a  = ga[f];
    float var = m2 - (2.0f * a - a * a) * m * m;   // == mean((h - a*m)^2)
    float sub = h[gid] - a * m;
    h[gid] = gg[f] * sub * rsqrtf(var + EPS) + gb[f];
}

// ---------------------------------------------------------------- mean pool + classifier
__global__ __launch_bounds__(256) void k_pool(const float* __restrict__ h,
                                              const int* __restrict__ ng,
                                              float* __restrict__ hg) {
    int gid = blockIdx.x * 256 + threadIdx.x;
    if (gid >= NN * HH) return;
    int n = gid >> 7, f = gid & 127;
    atomicAdd(&hg[ng[n] * HH + f], h[gid]);
}
__global__ __launch_bounds__(32) void k_cls(const float* __restrict__ hg,
                                            const float* __restrict__ cnt,
                                            const float* __restrict__ w,
                                            const float* __restrict__ b,
                                            float* __restrict__ out) {
    int g = blockIdx.x;
    int c = threadIdx.x;
    if (c >= CC) return;
    float ic = 1.0f / fmaxf(cnt[g], 1.0f);
    float s = b[c];
    #pragma unroll 4
    for (int k = 0; k < HH; ++k)
        s += hg[g * HH + k] * ic * w[k * CC + c];
    out[g * CC + c] = s;
}

// ================================================================ host
extern "C" void kernel_launch(void* const* d_in, const int* in_sizes, int n_in,
                              void* d_out, int out_size, void* d_ws, size_t ws_size,
                              hipStream_t stream) {
    const float* feat = (const float*)d_in[0];
    const int*   esrc = (const int*)d_in[1];
    const int*   edst = (const int*)d_in[2];
    const int*   ng   = (const int*)d_in[3];
    // d_in[4] = num_graphs (compile-time GG)
    const float* ln_g = (const float*)d_in[5];
    const float* ln_b = (const float*)d_in[6];
    const float* ws0  = (const float*)d_in[7];
    const float* wn0  = (const float*)d_in[8];
    const float* b0   = (const float*)d_in[9];
    const float* gg0  = (const float*)d_in[10];
    const float* gb0  = (const float*)d_in[11];
    const float* ga0  = (const float*)d_in[12];
    const float* ws1  = (const float*)d_in[13];
    const float* wn1  = (const float*)d_in[14];
    const float* b1   = (const float*)d_in[15];
    const float* gg1  = (const float*)d_in[16];
    const float* gb1  = (const float*)d_in[17];
    const float* ga1  = (const float*)d_in[18];
    const float* clw  = (const float*)d_in[19];
    const float* clb  = (const float*)d_in[20];
    float* out = (float*)d_out;

    // -------- workspace partition (h0 / agg padded to NPAD rows for the GEMM epilogue)
    char* p = (char*)d_ws;
    auto take = [&](size_t bytes) -> char* {
        char* r = p;
        p += (bytes + 255) & ~(size_t)255;
        return r;
    };
    float*  hln   = (float*) take((size_t)NN * 32 * 4);
    float*  deg   = (float*) take((size_t)NN * 4);
    float*  cnt   = (float*) take((size_t)GG * 4);
    float*  agg   = (float*) take((size_t)NPAD * HH * 4);
    __bf16* A0    = (__bf16*)take((size_t)NPAD * 64 * 2);
    __bf16* A1    = (__bf16*)take((size_t)NPAD * 256 * 2);
    float*  h0    = (float*) take((size_t)NPAD * HH * 4);
    float*  h1    = agg;                           // alias: agg is dead after pack1
    __bf16* WT0   = (__bf16*)take((size_t)HH * 64 * 2);
    __bf16* WT1   = (__bf16*)take((size_t)HH * 256 * 2);
    float*  gsum  = (float*) take((size_t)GG * HH * 4);
    float*  gsum2 = (float*) take((size_t)GG * HH * 4);
    float*  hg    = (float*) take((size_t)GG * HH * 4);

    const int nblk128 = (NN * HH + 255) / 256;     // 50000

    // -------- degree, counts, weights, input LN
    hipMemsetAsync(deg, 0, (size_t)NN * 4, stream);
    hipMemsetAsync(cnt, 0, (size_t)GG * 4, stream);
    k_deg<<<(EE + 255) / 256, 256, 0, stream>>>(edst, deg);
    k_cnt<<<(NN + 255) / 256, 256, 0, stream>>>(ng, cnt);
    k_prep_w0<<<(HH * 64) / 256, 256, 0, stream>>>(ws0, wn0, WT0);
    k_prep_w1<<<(HH * 256) / 256, 256, 0, stream>>>(ws1, wn1, WT1);
    k_ln<<<(NN + 255) / 256, 256, 0, stream>>>(feat, ln_g, ln_b, hln);

    // -------- layer 0
    hipMemsetAsync(agg, 0, (size_t)NN * 32 * 4, stream);
    k_edge_agg<5, 32><<<(EE * 5 + 255) / 256, 256, 0, stream>>>(esrc, edst, hln, agg);
    k_pack0<<<((size_t)NPAD * 64) / 256, 256, 0, stream>>>(hln, agg, deg, A0);
    sage_gemm_kernel<2><<<NPAD / 64, 256, 64 * (64 + 8) * 2, stream>>>(A0, WT0, b0, h0);
    hipMemsetAsync(gsum,  0, (size_t)GG * HH * 4, stream);
    hipMemsetAsync(gsum2, 0, (size_t)GG * HH * 4, stream);
    k_gn_sum<<<nblk128, 256, 0, stream>>>(h0, ng, gsum, gsum2);
    k_gn_apply<<<nblk128, 256, 0, stream>>>(h0, ng, cnt, gsum, gsum2, gg0, gb0, ga0);

    // -------- layer 1
    hipMemsetAsync(agg, 0, (size_t)NN * HH * 4, stream);
    k_edge_agg<32, 128><<<(EE * 32 + 255) / 256, 256, 0, stream>>>(esrc, edst, h0, agg);
    k_pack1<<<((size_t)NPAD * 256) / 256, 256, 0, stream>>>(h0, agg, deg, A1);
    sage_gemm_kernel<8><<<NPAD / 64, 256, 64 * (256 + 8) * 2, stream>>>(A1, WT1, b1, h1);
    hipMemsetAsync(gsum,  0, (size_t)GG * HH * 4, stream);
    hipMemsetAsync(gsum2, 0, (size_t)GG * HH * 4, stream);
    k_gn_sum<<<nblk128, 256, 0, stream>>>(h1, ng, gsum, gsum2);
    k_gn_apply<<<nblk128, 256, 0, stream>>>(h1, ng, cnt, gsum, gsum2, gg1, gb1, ga1);

    // -------- pool + classifier
    hipMemsetAsync(hg, 0, (size_t)GG * HH * 4, stream);
    k_pool<<<nblk128, 256, 0, stream>>>(h1, ng, hg);
    k_cls<<<GG, 32, 0, stream>>>(hg, cnt, clw, clb, out);

    (void)in_sizes; (void)n_in; (void)out_size; (void)ws_size;
}
